// PatchAlign3DEncoder_37598143709559
// MI455X (gfx1250) — matrix operand order repaired and checked by reading, not compiled, not run on hardware
//
#include <hip/hip_runtime.h>
#include <hip/hip_bf16.h>
#include <math.h>
#include <stdint.h>

// ---------------- problem constants ----------------
#define Bc    32
#define Np    8192
#define Gc    128
#define Mc    32
#define ENCc  256
#define DIMc  384
#define DEPTHc 12
#define HEADSc 6
#define HDc   64
#define Sc    129                 // 1 + Gc tokens
#define ROWS_T (Bc * Sc)          // 4128 (multiple of 32)
#define NGRP  (Bc * Gc)           // 4096
#define LN_EPSf 1e-5f
#define BN_EPSf 1e-5f

typedef __bf16 bf16;
typedef __attribute__((ext_vector_type(16))) __bf16 v16bf;
typedef __attribute__((ext_vector_type(8)))  float  v8f;

struct BfPair { bf16 lo, hi; };

// ---------------- WMMA fragment helpers ----------------
// A fragment: 16x32 bf16, row-major source, lane = M (lane&15), K split per ISA table
__device__ __forceinline__ v16bf load_frag_a(const bf16* base, int ldk, int lane) {
  int l15 = lane & 15, hl = lane >> 4;
  const bf16* row = base + (size_t)l15 * ldk;
  v16bf a;
#pragma unroll
  for (int i = 0; i < 8; ++i) {
    int k = ((i & 4) ? 16 : 0) + 8 * hl + 2 * (i & 3);
    BfPair p = *(const BfPair*)(row + k);
    a[2 * i] = p.lo; a[2 * i + 1] = p.hi;
  }
  return a;
}
// B fragment: 32x16 bf16 taken from transposed weights (N x K row-major), lane = N
__device__ __forceinline__ v16bf load_frag_b(const bf16* base, int ldk, int lane) {
  int l15 = lane & 15, hl = lane >> 4;
  const bf16* row = base + (size_t)l15 * ldk;
  v16bf b;
#pragma unroll
  for (int v = 0; v < 8; ++v) {
    int k = ((v & 4) ? 16 : 0) + 8 * hl + 2 * (v & 3);
    BfPair p = *(const BfPair*)(row + k);
    b[2 * v] = p.lo; b[2 * v + 1] = p.hi;
  }
  return b;
}
__device__ __forceinline__ v8f wmma_bf16(v16bf a, v16bf b, v8f c) {
  return __builtin_amdgcn_wmma_f32_16x16x32_bf16(false, a, false, b, (short)0, c, false, false);
}
__device__ __forceinline__ float gelu_exact(float x) {
  return 0.5f * x * (1.0f + erff(x * 0.70710678118654752f));
}
// low 32 bits of a generic pointer to LDS == workgroup-relative LDS byte address (ISA 10.2)
__device__ __forceinline__ uint32_t lds_addr32(const void* p) {
  return (uint32_t)(uintptr_t)p;
}

// ---------------- weight convert: (L,K,N) f32 -> (L,N,K) bf16 ----------------
__global__ void __launch_bounds__(256) convt_kernel(const float* __restrict__ in,
                                                    bf16* __restrict__ out,
                                                    int K, int N, long total) {
  long stride = (long)gridDim.x * 256;
  for (long idx = (long)blockIdx.x * 256 + threadIdx.x; idx < total; idx += stride) {
    long kn = (long)K * N;
    long l = idx / kn;
    long r = idx - l * kn;
    int k = (int)(r / N), n = (int)(r % N);
    out[l * kn + (long)n * K + k] = (bf16)in[idx];
  }
}

// ---------------- FPS: one block per batch ----------------
__global__ void __launch_bounds__(256) fps_kernel(const float* __restrict__ pts,
                                                  int* __restrict__ cidx,
                                                  float* __restrict__ center) {
  int b = blockIdx.x, tid = threadIdx.x;
  __shared__ float dist[Np];
  __shared__ float rv[256];
  __shared__ int   ri[256];
  __shared__ float cur[3];
  const float* px = pts + (size_t)b * 3 * Np;
  if (tid == 0) {
    cur[0] = px[0]; cur[1] = px[Np]; cur[2] = px[2 * Np];
    cidx[b * Gc] = 0;
    center[(size_t)(b * Gc) * 3 + 0] = cur[0];
    center[(size_t)(b * Gc) * 3 + 1] = cur[1];
    center[(size_t)(b * Gc) * 3 + 2] = cur[2];
  }
  __syncthreads();
  for (int n = tid; n < Np; n += 256) {
    float dx = px[n] - cur[0], dy = px[Np + n] - cur[1], dz = px[2 * Np + n] - cur[2];
    dist[n] = dx * dx + dy * dy + dz * dz;
  }
  for (int it = 1; it < Gc; ++it) {
    __syncthreads();
    float bv = -1.0f; int bi = 0x7fffffff;
    for (int n = tid; n < Np; n += 256) {
      float d = dist[n];
      if (d > bv) { bv = d; bi = n; }
    }
    rv[tid] = bv; ri[tid] = bi;
    __syncthreads();
    for (int st = 128; st > 0; st >>= 1) {
      if (tid < st) {
        float ov = rv[tid + st]; int oi = ri[tid + st];
        if (ov > rv[tid] || (ov == rv[tid] && oi < ri[tid])) { rv[tid] = ov; ri[tid] = oi; }
      }
      __syncthreads();
    }
    if (tid == 0) {
      int k = ri[0];
      cidx[b * Gc + it] = k;
      cur[0] = px[k]; cur[1] = px[Np + k]; cur[2] = px[2 * Np + k];
      center[(size_t)(b * Gc + it) * 3 + 0] = cur[0];
      center[(size_t)(b * Gc + it) * 3 + 1] = cur[1];
      center[(size_t)(b * Gc + it) * 3 + 2] = cur[2];
    }
    __syncthreads();
    for (int n = tid; n < Np; n += 256) {
      float dx = px[n] - cur[0], dy = px[Np + n] - cur[1], dz = px[2 * Np + n] - cur[2];
      float d = dx * dx + dy * dy + dz * dz;
      dist[n] = fminf(dist[n], d);
    }
  }
}

// ---------------- kNN: one block per (b,g); 32 iterative argmins ----------------
__global__ void __launch_bounds__(256) knn_kernel(const float* __restrict__ pts,
                                                  const float* __restrict__ center,
                                                  float* __restrict__ neigh) {
  int bg = blockIdx.x, tid = threadIdx.x;
  int b = bg >> 7;
  __shared__ float dist[Np];
  __shared__ float rv[256];
  __shared__ int   ri[256];
  const float* px = pts + (size_t)b * 3 * Np;
  float cx = center[(size_t)bg * 3 + 0];
  float cy = center[(size_t)bg * 3 + 1];
  float cz = center[(size_t)bg * 3 + 2];
  for (int n = tid; n < Np; n += 256) {
    float dx = px[n] - cx, dy = px[Np + n] - cy, dz = px[2 * Np + n] - cz;
    dist[n] = dx * dx + dy * dy + dz * dz;
  }
  for (int m = 0; m < Mc; ++m) {
    __syncthreads();
    float bv = 3.0e38f; int bi = 0x7fffffff;
    for (int n = tid; n < Np; n += 256) {
      float d = dist[n];
      if (d < bv) { bv = d; bi = n; }
    }
    rv[tid] = bv; ri[tid] = bi;
    __syncthreads();
    for (int st = 128; st > 0; st >>= 1) {
      if (tid < st) {
        float ov = rv[tid + st]; int oi = ri[tid + st];
        if (ov < rv[tid] || (ov == rv[tid] && oi < ri[tid])) { rv[tid] = ov; ri[tid] = oi; }
      }
      __syncthreads();
    }
    if (tid == 0) {
      int k = ri[0];
      size_t o = ((size_t)bg * Mc + m) * 3;
      neigh[o + 0] = px[k] - cx;
      neigh[o + 1] = px[Np + k] - cy;
      neigh[o + 2] = px[2 * Np + k] - cz;
      dist[k] = 3.4e38f;
    }
  }
}

// ---------------- fused group encoder: one block (8 waves) per group ----------------
__global__ void __launch_bounds__(256) encoder_kernel(
    const float* __restrict__ neigh,
    const float* __restrict__ w1, const float* __restrict__ b1,
    const float* __restrict__ bn1g, const float* __restrict__ bn1b,
    const bf16* __restrict__ w2t, const float* __restrict__ b2,
    const bf16* __restrict__ w3t, const float* __restrict__ b3,
    const float* __restrict__ bn2g, const float* __restrict__ bn2b,
    const bf16* __restrict__ w4t, const float* __restrict__ b4,
    bf16* __restrict__ tokens) {
  // time-sliced LDS: f1[0,8K) f2[8K,40K) H[40K,72K) f3[0,32K) f4[32K,64K)
  __shared__ char smem[72 * 1024];
  bf16*  sF1 = (bf16*)smem;                   // 32x128
  float* sF2 = (float*)(smem + 8 * 1024);     // 32x256
  bf16*  sH  = (bf16*)(smem + 40 * 1024);     // 32x512
  bf16*  sF3 = (bf16*)smem;                   // 32x512
  float* sF4 = (float*)(smem + 32 * 1024);    // 32x256

  int grp = blockIdx.x, tid = threadIdx.x;
  int lane = tid & 31, wave = tid >> 5;
  int l15 = lane & 15, hl = lane >> 4;
  const float* nb = neigh + (size_t)grp * Mc * 3;
  const float bnS = rsqrtf(1.0f + BN_EPSf);

  // ---- stage 1: f1 = relu(bn1(pg @ W1 + b1)) : 32x128, K=3 (VALU) ----
  for (int o = tid; o < 32 * 128; o += 256) {
    int r = o >> 7, c = o & 127;
    float p0 = nb[r * 3 + 0], p1 = nb[r * 3 + 1], p2 = nb[r * 3 + 2];
    float v = p0 * w1[c] + p1 * w1[128 + c] + p2 * w1[256 + c] + b1[c];
    v = v * (bn1g[c] * bnS) + bn1b[c];
    sF1[o] = (bf16)fmaxf(v, 0.0f);
  }
  __syncthreads();

  // ---- stage 2: f2 = f1 @ W2 + b2 : 32x256, K=128 (WMMA, 32 tiles) ----
  for (int j = 0; j < 4; ++j) {
    int t = wave + 8 * j;
    int m0 = (t & 1) * 16, n0 = (t >> 1) * 16;
    v8f acc = {};
    for (int kb = 0; kb < 128; kb += 32) {
      v16bf a = load_frag_a(sF1 + m0 * 128 + kb, 128, lane);
      v16bf bb = load_frag_b(w2t + (size_t)n0 * 128 + kb, 128, lane);
      acc = wmma_bf16(a, bb, acc);
    }
    int n = n0 + l15;
    float bv = b2[n];
#pragma unroll
    for (int i = 0; i < 8; ++i) sF2[(m0 + i + 8 * hl) * 256 + n] = acc[i] + bv;
  }
  __syncthreads();

  // ---- stage 3: fg = max over M; h = [fg | f2] : 32x512 bf16 ----
  {
    int c = tid;  // 256 threads, 256 columns
    float mx = -3.4e38f;
    for (int r = 0; r < 32; ++r) mx = fmaxf(mx, sF2[r * 256 + c]);
    for (int r = 0; r < 32; ++r) {
      sH[r * 512 + c]       = (bf16)mx;
      sH[r * 512 + 256 + c] = (bf16)sF2[r * 256 + c];
    }
  }
  __syncthreads();

  // ---- stage 4: f3 = relu(bn2(h @ W3 + b3)) : 32x512, K=512 (WMMA, 64 tiles) ----
  for (int j = 0; j < 8; ++j) {
    int t = wave + 8 * j;
    int m0 = (t & 1) * 16, n0 = (t >> 1) * 16;
    v8f acc = {};
    for (int kb = 0; kb < 512; kb += 32) {
      v16bf a = load_frag_a(sH + m0 * 512 + kb, 512, lane);
      v16bf bb = load_frag_b(w3t + (size_t)n0 * 512 + kb, 512, lane);
      acc = wmma_bf16(a, bb, acc);
    }
    int n = n0 + l15;
    float sc = bn2g[n] * bnS;
    float sh = b3[n] * sc + bn2b[n];
#pragma unroll
    for (int i = 0; i < 8; ++i) {
      float v = fmaxf(acc[i] * sc + sh, 0.0f);
      sF3[(m0 + i + 8 * hl) * 512 + n] = (bf16)v;
    }
  }
  __syncthreads();

  // ---- stage 5: f4 = f3 @ W4 + b4 : 32x256, K=512 (WMMA, 32 tiles) ----
  for (int j = 0; j < 4; ++j) {
    int t = wave + 8 * j;
    int m0 = (t & 1) * 16, n0 = (t >> 1) * 16;
    v8f acc = {};
    for (int kb = 0; kb < 512; kb += 32) {
      v16bf a = load_frag_a(sF3 + m0 * 512 + kb, 512, lane);
      v16bf bb = load_frag_b(w4t + (size_t)n0 * 512 + kb, 512, lane);
      acc = wmma_bf16(a, bb, acc);
    }
    int n = n0 + l15;
    float bv = b4[n];
#pragma unroll
    for (int i = 0; i < 8; ++i) sF4[(m0 + i + 8 * hl) * 256 + n] = acc[i] + bv;
  }
  __syncthreads();

  // ---- stage 6: token = max over M ----
  {
    int c = tid;
    float mx = -3.4e38f;
    for (int r = 0; r < 32; ++r) mx = fmaxf(mx, sF4[r * 256 + c]);
    tokens[(size_t)grp * 256 + c] = (bf16)mx;
  }
}

// ---------------- generic bf16 WMMA GEMM: C = epi(A @ Bt^T) ----------------
// A: rows x K bf16 row-major (rows % 32 == 0, K % 64 == 0)
// Bt: N x K bf16 row-major (transposed weights), N % 128 == 0 via grid.y
// A panel (32 x 64 per k-tile) is staged cooperatively into LDS with
// GLOBAL_LOAD_ASYNC_TO_LDS_B128 (ASYNCcnt), double-buffered so the async DMA
// of tile i+1 overlaps WMMA on tile i.
// mode: 0 = none, 1 = +bias, 2 = +bias then GELU, 3 = +bias + residual
__global__ void __launch_bounds__(256) gemm_bf16_kernel(
    const bf16* __restrict__ A, const bf16* __restrict__ Bt,
    const float* __restrict__ bias, const float* __restrict__ res,
    float* __restrict__ outF, bf16* __restrict__ outB,
    int K, int N, int mode) {
  __shared__ bf16 sA[2 * 32 * 64];            // double-buffered 4KB A panels
  int tid = threadIdx.x;
  int lane = tid & 31, wave = tid >> 5;
  int l15 = lane & 15, hl = lane >> 4;
  int mBase = blockIdx.x * 32;
  int m0 = mBase + (wave & 1) * 16;
  int n0 = blockIdx.y * 128 + (wave >> 1) * 32;
  v8f acc0 = {}, acc1 = {};
  const bf16* b0row = Bt + (size_t)n0 * K;
  const bf16* b1row = Bt + (size_t)(n0 + 16) * K;

  // async staging setup: thread -> (row, 16B chunk) of the 32x64 bf16 panel
  int srow = tid >> 3, schunk = tid & 7;
  uint32_t sAbase = lds_addr32(&sA[0]);
  unsigned long long abase = (unsigned long long)(uintptr_t)A;
  auto stage = [&](int buf, int kb) {
    uint32_t ldsa = sAbase + (uint32_t)buf * (32 * 64 * 2) + (uint32_t)(srow * 128 + schunk * 16);
    uint32_t goff = (uint32_t)(((mBase + srow) * K + kb) * 2 + schunk * 16);
    asm volatile("global_load_async_to_lds_b128 %0, %1, %2"
                 :: "v"(ldsa), "v"(goff), "s"(abase) : "memory");
  };

  int cur = 0;
  stage(0, 0);
  for (int kb = 0; kb < K; kb += 64) {
    asm volatile("s_wait_asynccnt 0x0" ::: "memory");
    __syncthreads();
    if (kb + 64 < K) stage(cur ^ 1, kb + 64);
    const bf16* ap = sA + cur * (32 * 64) + (wave & 1) * 16 * 64;
#pragma unroll
    for (int kk = 0; kk < 64; kk += 32) {
      __builtin_prefetch(b0row + kb + kk + 256, 0, 1);   // global_prefetch_b8
      v16bf a  = load_frag_a(ap + kk, 64, lane);
      v16bf b0 = load_frag_b(b0row + kb + kk, K, lane);
      v16bf b1 = load_frag_b(b1row + kb + kk, K, lane);
      acc0 = wmma_bf16(a, b0, acc0);
      acc1 = wmma_bf16(a, b1, acc1);
    }
    __syncthreads();
    cur ^= 1;
  }
#pragma unroll
  for (int sub = 0; sub < 2; ++sub) {
    v8f acc = sub ? acc1 : acc0;
    int n = n0 + sub * 16 + l15;
    float bv = bias ? bias[n] : 0.0f;
#pragma unroll
    for (int i = 0; i < 8; ++i) {
      int m = m0 + i + 8 * hl;
      float v = acc[i] + bv;
      if (mode == 2) v = gelu_exact(v);
      if (mode == 3) v += res[(size_t)m * N + n];
      if (outF) outF[(size_t)m * N + n] = v;
      if (outB) outB[(size_t)m * N + n] = (bf16)v;
    }
  }
}

// ---------------- pos MLP stage 1: gelu(center @ pos1 + b), K=3 ----------------
__global__ void __launch_bounds__(256) pe1_kernel(const float* __restrict__ center,
                                                  const float* __restrict__ w,
                                                  const float* __restrict__ bias,
                                                  bf16* __restrict__ out) {
  int idx = blockIdx.x * 256 + threadIdx.x;
  if (idx >= NGRP * 128) return;
  int r = idx >> 7, c = idx & 127;
  float p0 = center[(size_t)r * 3 + 0], p1 = center[(size_t)r * 3 + 1], p2 = center[(size_t)r * 3 + 2];
  float v = p0 * w[c] + p1 * w[128 + c] + p2 * w[256 + c] + bias[c];
  out[idx] = (bf16)gelu_exact(v);
}

// ---------------- build x (cls | tokens) and pos (cls_pos | pe) ----------------
__global__ void __launch_bounds__(128) scatter_kernel(
    const float* __restrict__ tok384, const float* __restrict__ pe,
    const float* __restrict__ cls_tok, const float* __restrict__ cls_pos,
    float* __restrict__ x, float* __restrict__ pos) {
  int r = blockIdx.x, tid = threadIdx.x;
  int b = r / Sc, s = r % Sc;
#pragma unroll
  for (int j = 0; j < 3; ++j) {
    int c = tid + j * 128;
    float xv, pv;
    if (s == 0) { xv = cls_tok[c]; pv = cls_pos[c]; }
    else {
      size_t t = (size_t)(b * Gc + s - 1) * DIMc + c;
      xv = tok384[t]; pv = pe[t];
    }
    x[(size_t)r * DIMc + c] = xv;
    pos[(size_t)r * DIMc + c] = pv;
  }
}

// ---------------- fused (x [+pos]) + LayerNorm -> bf16 (and f32 residual copy) ----
__global__ void __launch_bounds__(128) add_ln_kernel(
    const float* __restrict__ xb, const float* __restrict__ pb,
    const float* __restrict__ g, const float* __restrict__ be,
    float* __restrict__ xout, bf16* __restrict__ hb) {
  int r = blockIdx.x, tid = threadIdx.x;
  __shared__ float red[128], red2[128];
  float v[3];
  float sum = 0.0f, sq = 0.0f;
#pragma unroll
  for (int j = 0; j < 3; ++j) {
    int c = tid + j * 128;
    float t = xb[(size_t)r * DIMc + c] + (pb ? pb[(size_t)r * DIMc + c] : 0.0f);
    v[j] = t;
    if (xout) xout[(size_t)r * DIMc + c] = t;
    sum += t; sq += t * t;
  }
  red[tid] = sum; red2[tid] = sq;
  __syncthreads();
  for (int st = 64; st > 0; st >>= 1) {
    if (tid < st) { red[tid] += red[tid + st]; red2[tid] += red2[tid + st]; }
    __syncthreads();
  }
  float mean = red[0] * (1.0f / DIMc);
  float var = red2[0] * (1.0f / DIMc) - mean * mean;
  float inv = rsqrtf(var + LN_EPSf);
#pragma unroll
  for (int j = 0; j < 3; ++j) {
    int c = tid + j * 128;
    hb[(size_t)r * DIMc + c] = (bf16)((v[j] - mean) * inv * g[c] + be[c]);
  }
}

// ---------------- attention: one block per (s, head, b) ----------------
__global__ void __launch_bounds__(128) attn_kernel(const float* __restrict__ qkv,
                                                   bf16* __restrict__ obf) {
  int s = blockIdx.x, h = blockIdx.y, b = blockIdx.z;
  int tid = threadIdx.x;
  __shared__ float sq[HDc];
  __shared__ float sc[Sc];
  __shared__ float red[128];
  const float* qrow = qkv + (size_t)(b * Sc + s) * (3 * DIMc) + h * HDc;
  if (tid < HDc) sq[tid] = qrow[tid];
  __syncthreads();
  float lmax = -3.4e38f;
  for (int t = tid; t < Sc; t += 128) {
    const float* krow = qkv + (size_t)(b * Sc + t) * (3 * DIMc) + DIMc + h * HDc;
    float acc = 0.0f;
#pragma unroll
    for (int d = 0; d < HDc; ++d) acc += sq[d] * krow[d];
    acc *= 0.125f;  // HD^-0.5
    sc[t] = acc;
    lmax = fmaxf(lmax, acc);
  }
  red[tid] = lmax;
  __syncthreads();
  for (int st = 64; st > 0; st >>= 1) {
    if (tid < st) red[tid] = fmaxf(red[tid], red[tid + st]);
    __syncthreads();
  }
  float mx = red[0];
  __syncthreads();
  float lsum = 0.0f;
  for (int t = tid; t < Sc; t += 128) {
    float e = __expf(sc[t] - mx);
    sc[t] = e;
    lsum += e;
  }
  red[tid] = lsum;
  __syncthreads();
  for (int st = 64; st > 0; st >>= 1) {
    if (tid < st) red[tid] += red[tid + st];
    __syncthreads();
  }
  float inv = 1.0f / red[0];
  for (int d = tid; d < HDc; d += 128) {
    float acc = 0.0f;
    for (int t = 0; t < Sc; ++t)
      acc += sc[t] * qkv[(size_t)(b * Sc + t) * (3 * DIMc) + 2 * DIMc + h * HDc + d];
    obf[(size_t)(b * Sc + s) * DIMc + h * HDc + d] = (bf16)(acc * inv);
  }
}

// ---------------- final LN + transpose into (B, DIM, G) output ----------------
__global__ void __launch_bounds__(128) final_ln_kernel(const float* __restrict__ x,
                                                       const float* __restrict__ g,
                                                       const float* __restrict__ be,
                                                       float* __restrict__ out) {
  int r = blockIdx.x;
  int b = r / Sc, s = r % Sc;
  if (s == 0) return;
  int tid = threadIdx.x;
  __shared__ float red[128], red2[128];
  float v[3];
  float sum = 0.0f, sq = 0.0f;
#pragma unroll
  for (int j = 0; j < 3; ++j) {
    int c = tid + j * 128;
    float t = x[(size_t)r * DIMc + c];
    v[j] = t; sum += t; sq += t * t;
  }
  red[tid] = sum; red2[tid] = sq;
  __syncthreads();
  for (int st = 64; st > 0; st >>= 1) {
    if (tid < st) { red[tid] += red[tid + st]; red2[tid] += red2[tid + st]; }
    __syncthreads();
  }
  float mean = red[0] * (1.0f / DIMc);
  float var = red2[0] * (1.0f / DIMc) - mean * mean;
  float inv = rsqrtf(var + LN_EPSf);
#pragma unroll
  for (int j = 0; j < 3; ++j) {
    int c = tid + j * 128;
    float val = (v[j] - mean) * inv * g[c] + be[c];
    out[(size_t)b * DIMc * Gc + (size_t)c * Gc + (s - 1)] = val;
  }
}

// ---------------- host side ----------------
static inline size_t alignup256(size_t x) { return (x + 255) & ~(size_t)255; }

static void gemm(hipStream_t s, const bf16* A, const bf16* Bt, const float* bias,
                 const float* res, float* outF, bf16* outB,
                 int rows, int K, int N, int mode) {
  dim3 grid(rows / 32, N / 128);
  gemm_bf16_kernel<<<grid, 256, 0, s>>>(A, Bt, bias, res, outF, outB, K, N, mode);
}
static void convt(hipStream_t s, const float* in, bf16* out, int K, int N, int L) {
  long total = (long)L * K * N;
  int grid = (int)((total + 255) / 256);
  convt_kernel<<<grid, 256, 0, s>>>(in, out, K, N, total);
}

extern "C" void kernel_launch(void* const* d_in, const int* in_sizes, int n_in,
                              void* d_out, int out_size, void* d_ws, size_t ws_size,
                              hipStream_t stream) {
  (void)in_sizes; (void)n_in; (void)out_size; (void)ws_size;
  const float* pts    = (const float*)d_in[0];
  const float* e1w    = (const float*)d_in[1];
  const float* e1b    = (const float*)d_in[2];
  const float* bn1g   = (const float*)d_in[3];
  const float* bn1b   = (const float*)d_in[4];
  const float* e2w    = (const float*)d_in[5];
  const float* e2b    = (const float*)d_in[6];
  const float* e3w    = (const float*)d_in[7];
  const float* e3b    = (const float*)d_in[8];
  const float* bn2g   = (const float*)d_in[9];
  const float* bn2b   = (const float*)d_in[10];
  const float* e4w    = (const float*)d_in[11];
  const float* e4b    = (const float*)d_in[12];
  const float* redw   = (const float*)d_in[13];
  const float* redb   = (const float*)d_in[14];
  const float* clstok = (const float*)d_in[15];
  const float* clspos = (const float*)d_in[16];
  const float* pos1w  = (const float*)d_in[17];
  const float* pos1b  = (const float*)d_in[18];
  const float* pos2w  = (const float*)d_in[19];
  const float* pos2b  = (const float*)d_in[20];
  const float* ln1g   = (const float*)d_in[21];
  const float* ln1b   = (const float*)d_in[22];
  const float* qkvw   = (const float*)d_in[23];
  const float* projw  = (const float*)d_in[24];
  const float* projb  = (const float*)d_in[25];
  const float* ln2g   = (const float*)d_in[26];
  const float* ln2b   = (const float*)d_in[27];
  const float* fc1w   = (const float*)d_in[28];
  const float* fc1b   = (const float*)d_in[29];
  const float* fc2w   = (const float*)d_in[30];
  const float* fc2b   = (const float*)d_in[31];
  const float* normg  = (const float*)d_in[32];
  const float* normb  = (const float*)d_in[33];

  char* ws = (char*)d_ws;
  size_t off = 0;
  auto take = [&](size_t bytes) -> char* {
    char* p = ws + off;
    off += alignup256(bytes);
    return p;
  };
  int*   cidx    = (int*)  take((size_t)NGRP * 4);
  float* center  = (float*)take((size_t)NGRP * 3 * 4);
  float* neigh   = (float*)take((size_t)NGRP * Mc * 3 * 4);
  bf16*  tokens  = (bf16*) take((size_t)NGRP * ENCc * 2);
  float* tok384  = (float*)take((size_t)NGRP * DIMc * 4);
  bf16*  pe1     = (bf16*) take((size_t)NGRP * 128 * 2);
  float* pe      = (float*)take((size_t)NGRP * DIMc * 4);
  float* x       = (float*)take((size_t)ROWS_T * DIMc * 4);
  float* xin     = (float*)take((size_t)ROWS_T * DIMc * 4);
  float* x1      = (float*)take((size_t)ROWS_T * DIMc * 4);
  float* posb    = (float*)take((size_t)ROWS_T * DIMc * 4);
  bf16*  hln     = (bf16*) take((size_t)ROWS_T * DIMc * 2);
  bf16*  g1      = (bf16*) take((size_t)ROWS_T * 4 * DIMc * 2);
  float* qkvb    = (float*)take((size_t)ROWS_T * 3 * DIMc * 4);
  bf16*  obf     = (bf16*) take((size_t)ROWS_T * DIMc * 2);
  bf16*  w_e2t   = (bf16*) take((size_t)256 * 128 * 2);
  bf16*  w_e3t   = (bf16*) take((size_t)512 * 512 * 2);
  bf16*  w_e4t   = (bf16*) take((size_t)256 * 512 * 2);
  bf16*  w_redt  = (bf16*) take((size_t)DIMc * ENCc * 2);
  bf16*  w_pos2t = (bf16*) take((size_t)DIMc * 128 * 2);
  bf16*  w_qkvt  = (bf16*) take((size_t)DEPTHc * 3 * DIMc * DIMc * 2);
  bf16*  w_projt = (bf16*) take((size_t)DEPTHc * DIMc * DIMc * 2);
  bf16*  w_fc1t  = (bf16*) take((size_t)DEPTHc * 4 * DIMc * DIMc * 2);
  bf16*  w_fc2t  = (bf16*) take((size_t)DEPTHc * DIMc * 4 * DIMc * 2);

  // weight conversion (f32 KxN -> bf16 NxK, per layer)
  convt(stream, e2w,   w_e2t,   128,  256,  1);
  convt(stream, e3w,   w_e3t,   512,  512,  1);
  convt(stream, e4w,   w_e4t,   512,  256,  1);
  convt(stream, redw,  w_redt,  256,  384,  1);
  convt(stream, pos2w, w_pos2t, 128,  384,  1);
  convt(stream, qkvw,  w_qkvt,  384,  1152, DEPTHc);
  convt(stream, projw, w_projt, 384,  384,  DEPTHc);
  convt(stream, fc1w,  w_fc1t,  384,  1536, DEPTHc);
  convt(stream, fc2w,  w_fc2t,  1536, 384,  DEPTHc);

  // geometry
  fps_kernel<<<Bc, 256, 0, stream>>>(pts, cidx, center);
  knn_kernel<<<NGRP, 256, 0, stream>>>(pts, center, neigh);

  // group encoder (fused WMMA PointNet)
  encoder_kernel<<<NGRP, 256, 0, stream>>>(neigh, e1w, e1b, bn1g, bn1b,
                                           w_e2t, e2b, w_e3t, e3b, bn2g, bn2b,
                                           w_e4t, e4b, tokens);
  // tokens @ red_w + red_b
  gemm(stream, tokens, w_redt, redb, nullptr, tok384, nullptr, NGRP, 256, 384, 1);
  // positional MLP
  pe1_kernel<<<(NGRP * 128) / 256, 256, 0, stream>>>(center, pos1w, pos1b, pe1);
  gemm(stream, pe1, w_pos2t, pos2b, nullptr, pe, nullptr, NGRP, 128, 384, 1);
  // assemble x and pos (with cls rows)
  scatter_kernel<<<ROWS_T, 128, 0, stream>>>(tok384, pe, clstok, clspos, x, posb);

  // transformer
  for (int L = 0; L < DEPTHc; ++L) {
    const bf16* qkvt = w_qkvt + (size_t)L * 3 * DIMc * DIMc;
    const bf16* prjt = w_projt + (size_t)L * DIMc * DIMc;
    const bf16* f1t  = w_fc1t + (size_t)L * 4 * DIMc * DIMc;
    const bf16* f2t  = w_fc2t + (size_t)L * DIMc * 4 * DIMc;
    // xin = x + pos ; h = LN1(xin)
    add_ln_kernel<<<ROWS_T, 128, 0, stream>>>(x, posb, ln1g + L * DIMc, ln1b + L * DIMc, xin, hln);
    // qkv = h @ Wqkv (no bias)
    gemm(stream, hln, qkvt, nullptr, nullptr, qkvb, nullptr, ROWS_T, DIMc, 3 * DIMc, 0);
    // attention -> bf16
    attn_kernel<<<dim3(Sc, HEADSc, Bc), 128, 0, stream>>>(qkvb, obf);
    // x1 = xin + o @ Wp + bp
    gemm(stream, obf, prjt, projb + L * DIMc, xin, x1, nullptr, ROWS_T, DIMc, DIMc, 3);
    // h2 = LN2(x1)
    add_ln_kernel<<<ROWS_T, 128, 0, stream>>>(x1, nullptr, ln2g + L * DIMc, ln2b + L * DIMc, nullptr, hln);
    // g1 = gelu(h2 @ W1 + b1) -> bf16
    gemm(stream, hln, f1t, fc1b + L * 4 * DIMc, nullptr, nullptr, g1, ROWS_T, DIMc, 4 * DIMc, 2);
    // x = x1 + g1 @ W2 + b2
    gemm(stream, g1, f2t, fc2b + L * DIMc, x1, x, nullptr, ROWS_T, 4 * DIMc, DIMc, 3);
  }

  // final LN + transpose into (B, DIM, G)
  final_ln_kernel<<<ROWS_T, 128, 0, stream>>>(x, normg, normb, (float*)d_out);
}
